// Encoder_36369783063167
// MI455X (gfx1250) — matrix operand order repaired and checked by reading, not compiled, not run on hardware
//
#include <hip/hip_runtime.h>
#include <hip/hip_bf16.h>
#include <math.h>

typedef __attribute__((ext_vector_type(16))) __bf16 v16bf;
typedef __attribute__((ext_vector_type(8)))  float  v8f;

#define D     128   // D_IN == D_HID; also combined (mu|logvar) output width
#define SSTR  136   // LDS row stride in bf16 (pad 128->136: bank-conflict-free halves)

union BF16x16 { v16bf v; __bf16 e[16]; };
union LD16    { uint4 q[2]; v16bf v; };

// ---------------- utility: zero a buffer (graph-capture safe) ----------------
__global__ void zero_f4(float4* __restrict__ p, int n4) {
  int i = blockIdx.x * blockDim.x + threadIdx.x;
  if (i < n4) p[i] = make_float4(0.f, 0.f, 0.f, 0.f);
}

// ------------- weight prep: fp32 -> bf16, build combined mu|lv ---------------
__global__ void prep_weights(const float* __restrict__ W1rel,
                             const float* __restrict__ W1root,
                             const float* __restrict__ Wmurel,
                             const float* __restrict__ Wmuroot,
                             const float* __restrict__ Wlvrel,
                             const float* __restrict__ Wlvroot,
                             const float* __restrict__ bmu,
                             const float* __restrict__ blv,
                             __bf16* __restrict__ w1rel,
                             __bf16* __restrict__ w1root,
                             __bf16* __restrict__ w2rel,
                             __bf16* __restrict__ w2root,
                             float* __restrict__ bias2) {
  int i = blockIdx.x * blockDim.x + threadIdx.x;
  if (i < D * D) {
    w1rel[i]  = (__bf16)W1rel[i];
    w1root[i] = (__bf16)W1root[i];
    int row = i >> 7, col = i & 127;
    float vr = (row < 64) ? Wmurel[row * D + col]  : Wlvrel[(row - 64) * D + col];
    float vt = (row < 64) ? Wmuroot[row * D + col] : Wlvroot[(row - 64) * D + col];
    w2rel[i]  = (__bf16)vr;
    w2root[i] = (__bf16)vt;
  }
  if (i < D) bias2[i] = (i < 64) ? bmu[i] : blv[i - 64];
}

// ---------------- degree count + reciprocal ----------------------------------
__global__ void count_deg(const int* __restrict__ dst, float* __restrict__ cnt, int e) {
  int i = blockIdx.x * blockDim.x + threadIdx.x;
  if (i < e) atomicAdd(&cnt[dst[i]], 1.0f);
}

__global__ void make_inv(const float* __restrict__ cnt, float* __restrict__ inv, int n) {
  int i = blockIdx.x * blockDim.x + threadIdx.x;
  if (i < n) inv[i] = 1.0f / fmaxf(cnt[i], 1.0f);
}

// --------- edge scatter: one wave32 per edge, float4 per lane (128 f32) ------
__global__ __launch_bounds__(256)
void scatter_edges(const float* __restrict__ x,
                   const int* __restrict__ src,
                   const int* __restrict__ dst,
                   const float* __restrict__ ew,
                   float* __restrict__ agg, int e) {
  int gid  = blockIdx.x * blockDim.x + threadIdx.x;
  int wid  = gid >> 5;
  int lane = threadIdx.x & 31;
  if (wid >= e) return;
  int   s = src[wid];
  int   d = dst[wid];
  float w = ew[wid];
  float4 v = ((const float4*)(x + (long)s * D))[lane];     // 32 lanes * 4 = 128
  float* a = agg + (long)d * D + lane * 4;
  atomicAdd(a + 0, v.x * w);
  atomicAdd(a + 1, v.y * w);
  atomicAdd(a + 2, v.z * w);
  atomicAdd(a + 3, v.w * w);
}

// ---- fused SAGE GEMM: out = (agg*inv) @ Wrel^T + xin @ Wroot^T + bias -------
// One wave = one 16-row tile, loops 8 column tiles. Both weight matrices are
// staged in LDS once per block (padded stride SSTR for bank spread), then read
// with ds_load_b128 pairs. Two independent WMMA accumulator chains (rel/root).
//
// A layout (16-bit, 16x32): lane L, row = L%16; lanes 0-15 hold K 0-15,
// lanes 16-31 hold K 16-31 per 32-wide k-step.
// B layout (16-bit, 32x16): lane L, col = L%16; lanes 0-15 K 0-15, lanes 16-31
// K 16-31 -> row-major W[o,k] gives a contiguous 32B read per lane.
__device__ inline void pack8(BF16x16& o, int at, float4 a, float4 b, float s) {
  o.e[at + 0] = (__bf16)(a.x * s); o.e[at + 1] = (__bf16)(a.y * s);
  o.e[at + 2] = (__bf16)(a.z * s); o.e[at + 3] = (__bf16)(a.w * s);
  o.e[at + 4] = (__bf16)(b.x * s); o.e[at + 5] = (__bf16)(b.y * s);
  o.e[at + 6] = (__bf16)(b.z * s); o.e[at + 7] = (__bf16)(b.w * s);
}

template <int LAYER>
__global__ __launch_bounds__(256, 1)
void fused_sage_gemm(const float* __restrict__ agg,     // pre-division aggregate [n,128]
                     const float* __restrict__ inv,     // [n]
                     const float* __restrict__ xin,     // root-path input [n,128]
                     const __bf16* __restrict__ wrel,   // [128,128] bf16 row-major (o,k)
                     const __bf16* __restrict__ wroot,  // [128,128]
                     const float* __restrict__ bias,    // [128]
                     float* __restrict__ out_h,         // LAYER==1: h [n,128] (ELU)
                     float* __restrict__ out_mu,        // LAYER==2: mu [n,64]
                     float* __restrict__ out_lv,        // LAYER==2: logvar [n,64]
                     int n_tiles) {
  extern __shared__ char smem[];                 // 2 * 128 * SSTR bf16 = 68 KB
  __bf16* sW = (__bf16*)smem;

  // ---- cooperative weight staging: one 256B row per thread -------------------
  {
    int t   = threadIdx.x;                       // 0..255
    int mat = t >> 7;                            // 0 = rel, 1 = root
    int row = t & 127;
    const uint4* g = (const uint4*)((mat ? wroot : wrel) + (size_t)row * D);
    uint4* s = (uint4*)(sW + (size_t)(mat * D + row) * SSTR);
#pragma unroll
    for (int q = 0; q < 16; ++q) s[q] = g[q];    // 16 x b128: global -> LDS
  }
  __syncthreads();

  int wave = threadIdx.x >> 5;
  int lane = threadIdx.x & 31;
  int tile = blockIdx.x * 8 + wave;
  if (tile >= n_tiles) return;                   // wave-uniform: EXEC stays all-1

  int  mrow = lane & 15;
  int  half = lane >> 4;
  long r    = (long)tile * 16 + mrow;
  float sc  = inv[r];
  const float* pm = agg + r * D;
  const float* px = xin + r * D;

  // Stage both A operands (mean and root) for all 4 k-steps in registers.
  BF16x16 Am[4], Ax[4];
#pragma unroll
  for (int k = 0; k < 4; ++k) {
    int b0 = k * 32 + half * 16;                 // K chunk 1
    int b1 = b0 + 8;                             // K chunk 2
    pack8(Am[k], 0, *(const float4*)(pm + b0), *(const float4*)(pm + b0 + 4), sc);
    pack8(Am[k], 8, *(const float4*)(pm + b1), *(const float4*)(pm + b1 + 4), sc);
    pack8(Ax[k], 0, *(const float4*)(px + b0), *(const float4*)(px + b0 + 4), 1.0f);
    pack8(Ax[k], 8, *(const float4*)(px + b1), *(const float4*)(px + b1 + 4), 1.0f);
  }

#pragma unroll 1
  for (int j = 0; j < 8; ++j) {
    v8f accA = {0.f, 0.f, 0.f, 0.f, 0.f, 0.f, 0.f, 0.f};   // rel chain
    v8f accB = {0.f, 0.f, 0.f, 0.f, 0.f, 0.f, 0.f, 0.f};   // root chain
    const __bf16* srel  = sW + (size_t)(j * 16 + mrow) * SSTR + half * 16;
    const __bf16* sroot = sW + (size_t)(D + j * 16 + mrow) * SSTR + half * 16;
#pragma unroll
    for (int k = 0; k < 4; ++k) {
      LD16 brel, broot;
      const uint4* p0 = (const uint4*)(srel + k * 32);     // ds_load_b128 x2
      brel.q[0] = p0[0]; brel.q[1] = p0[1];
      const uint4* p1 = (const uint4*)(sroot + k * 32);
      broot.q[0] = p1[0]; broot.q[1] = p1[1];
      accA = __builtin_amdgcn_wmma_f32_16x16x32_bf16(
          false, Am[k].v, false, brel.v, (short)0, accA, false, false);
      accB = __builtin_amdgcn_wmma_f32_16x16x32_bf16(
          false, Ax[k].v, false, broot.v, (short)0, accB, false, false);
    }

    int   col  = j * 16 + mrow;
    float bcol = bias[col];
    if constexpr (LAYER == 1) {
#pragma unroll
      for (int i = 0; i < 8; ++i) {              // acc[i]: row = i + half*8, col = mrow
        long row = (long)tile * 16 + i + half * 8;
        float v = accA[i] + accB[i] + bcol;
        v = v > 0.f ? v : (__expf(v) - 1.f);     // ELU
        out_h[row * D + col] = v;
      }
    } else {
      // cols 0-63 -> mu, 64-127 -> logvar; j<4 is wave-uniform
      float* outp = (j < 4) ? (out_mu + col) : (out_lv + (col - 64));
#pragma unroll
      for (int i = 0; i < 8; ++i) {
        long row = (long)tile * 16 + i + half * 8;
        outp[row * 64] = accA[i] + accB[i] + bcol;
      }
    }
  }
}

// -----------------------------------------------------------------------------
extern "C" void kernel_launch(void* const* d_in, const int* in_sizes, int n_in,
                              void* d_out, int out_size, void* d_ws, size_t ws_size,
                              hipStream_t stream) {
  const float* x      = (const float*)d_in[0];
  const int*   eidx   = (const int*)  d_in[1];
  const float* ew     = (const float*)d_in[2];
  const float* W1rel  = (const float*)d_in[3];
  const float* b1     = (const float*)d_in[4];
  const float* W1root = (const float*)d_in[5];
  const float* Wmurel = (const float*)d_in[6];
  const float* bmu    = (const float*)d_in[7];
  const float* Wmuroot= (const float*)d_in[8];
  const float* Wlvrel = (const float*)d_in[9];
  const float* blv    = (const float*)d_in[10];
  const float* Wlvroot= (const float*)d_in[11];

  const int n = in_sizes[0] / D;    // 50000
  const int e = in_sizes[2];        // 800000
  const int* src = eidx;
  const int* dst = eidx + e;

  // ---- workspace carve-out (256B aligned slices) ----
  char*  ws  = (char*)d_ws;
  size_t off = 0;
  auto take = [&](size_t bytes) -> void* {
    void* p = ws + off;
    off += (bytes + 255) & ~(size_t)255;
    return p;
  };
  float*  cnt    = (float*) take((size_t)n * 4);
  float*  inv    = (float*) take((size_t)n * 4);
  float*  agg1   = (float*) take((size_t)n * D * 4);
  float*  agg2   = (float*) take((size_t)n * D * 4);
  float*  h      = (float*) take((size_t)n * D * 4);
  __bf16* w1rel  = (__bf16*)take((size_t)D * D * 2);
  __bf16* w1root = (__bf16*)take((size_t)D * D * 2);
  __bf16* w2rel  = (__bf16*)take((size_t)D * D * 2);
  __bf16* w2root = (__bf16*)take((size_t)D * D * 2);
  float*  bias2  = (float*) take((size_t)D * 4);

  float* mu = (float*)d_out;
  float* lv = (float*)d_out + (size_t)n * 64;

  const int tpb = 256;

  // 1) zero accumulation buffers (padding-safe: slices are 256B padded)
  {
    int n4 = (n + 3) / 4;
    zero_f4<<<(n4 + tpb - 1) / tpb, tpb, 0, stream>>>((float4*)cnt, n4);
    int m4 = (n * D) / 4;
    zero_f4<<<(m4 + tpb - 1) / tpb, tpb, 0, stream>>>((float4*)agg1, m4);
    zero_f4<<<(m4 + tpb - 1) / tpb, tpb, 0, stream>>>((float4*)agg2, m4);
  }

  // 2) weights -> bf16 (+ combined mu|logvar weights & bias)
  prep_weights<<<(D * D + tpb - 1) / tpb, tpb, 0, stream>>>(
      W1rel, W1root, Wmurel, Wmuroot, Wlvrel, Wlvroot, bmu, blv,
      w1rel, w1root, w2rel, w2root, bias2);

  // 3) degree counts and reciprocal (shared by all three convs)
  count_deg<<<(e + tpb - 1) / tpb, tpb, 0, stream>>>(dst, cnt, e);
  make_inv<<<(n + tpb - 1) / tpb, tpb, 0, stream>>>(cnt, inv, n);

  // 4) layer-1 aggregation: agg1 += ew * x[src]
  {
    long threads = (long)e * 32;
    scatter_edges<<<(unsigned)((threads + tpb - 1) / tpb), tpb, 0, stream>>>(
        x, src, dst, ew, agg1, e);
  }

  // 5) h = ELU(mean1 @ W1rel^T + b1 + x @ W1root^T)
  const int n_tiles   = n / 16;                  // 3125 (exact for N=50000)
  const int gblocks   = (n_tiles + 7) / 8;
  const size_t smembytes = (size_t)2 * D * SSTR * sizeof(__bf16);  // 68 KB dynamic LDS
  fused_sage_gemm<1><<<gblocks, tpb, smembytes, stream>>>(
      agg1, inv, x, w1rel, w1root, b1, h, nullptr, nullptr, n_tiles);

  // 6) shared aggregation for conv_mu AND conv_logvar (computed once)
  {
    long threads = (long)e * 32;
    scatter_edges<<<(unsigned)((threads + tpb - 1) / tpb), tpb, 0, stream>>>(
        h, src, dst, ew, agg2, e);
  }

  // 7) [mu | logvar] = mean2 @ W2rel^T + bias2 + h @ W2root^T
  fused_sage_gemm<2><<<gblocks, tpb, smembytes, stream>>>(
      agg2, inv, h, w2rel, w2root, bias2, nullptr, mu, lv, n_tiles);
}
